// MetapathEmbed_73882027425809
// MI455X (gfx1250) — matrix operand order repaired and checked by reading, not compiled, not run on hardware
//
#include <hip/hip_runtime.h>
#include <cstddef>

#define N_CARDS 16384
#define N_PATHS 16384
#define D_IN    128
#define D_MP    128
#define BATCHSZ 4096
#define NNZ_TOT 262144

typedef __attribute__((ext_vector_type(2)))  float  v2f;
typedef __attribute__((ext_vector_type(8)))  float  v8f;
typedef __attribute__((ext_vector_type(8)))  __bf16 v8bf;
typedef __attribute__((ext_vector_type(16))) __bf16 v16bf;

static __device__ __forceinline__ v8f wmma_f32_4(v2f a, v2f b, v8f c) {
  return __builtin_amdgcn_wmma_f32_16x16x4_f32(
      false, a, false, b, (short)0, c, false, false);
}

static __device__ __forceinline__ v8f wmma_bf16_32(v16bf a, v16bf b, v8f c) {
  return __builtin_amdgcn_wmma_f32_16x16x32_bf16(
      false, a, false, b, (short)0, c, false, false);
}

static __device__ __forceinline__ void bf16_split(float x, __bf16& hi, __bf16& lo) {
  hi = (__bf16)x;                 // RNE
  lo = (__bf16)(x - (float)hi);   // subtraction exact in fp32
}

static __device__ __forceinline__ v16bf cat8(v8bf a, v8bf b) {
  return __builtin_shufflevector(a, b, 0, 1, 2, 3, 4, 5, 6, 7,
                                 8, 9, 10, 11, 12, 13, 14, 15);
}

// ---------------------------------------------------------------------------
// Kernel 1: transformed = swish(E @ W + b), emitted as pre-split, transposed
// bf16 pair:  tf_hiT/tf_loT [D_MP][N_CARDS].
// grid = N_CARDS/128 blocks of 256 threads (8 waves); wave w -> rows w*16..+15
// ---------------------------------------------------------------------------
__global__ __launch_bounds__(256) void mpe_transform_swish(
    const float* __restrict__ E, const float* __restrict__ W,
    const float* __restrict__ b, __bf16* __restrict__ tf_hiT,
    __bf16* __restrict__ tf_loT) {
  __shared__ float Wl[D_IN * D_MP];  // 64 KB
  __shared__ float bl[D_MP];
  const int tid = threadIdx.x;
  {
    const float4* src = (const float4*)W;
    float4* dst = (float4*)Wl;
    for (int i = tid; i < D_IN * D_MP / 4; i += 256) dst[i] = src[i];
    if (tid < D_MP) bl[tid] = b[tid];
  }
  __syncthreads();

  const int wave  = tid >> 5;
  const int lane  = tid & 31;
  const int lhalf = lane >> 4;
  const int lmod  = lane & 15;
  const int m0 = blockIdx.x * 128 + wave * 16;

  // A 16x4 f32 layout: lane l holds M = l&15; VGPR v holds K = (l>>4)*2 + v.
  v2f afrag[32];
  const float2* erow = (const float2*)(E + (size_t)(m0 + lmod) * D_IN);
  #pragma unroll
  for (int kk = 0; kk < 32; ++kk) {
    float2 t = erow[kk * 2 + lhalf];
    afrag[kk] = (v2f){t.x, t.y};
  }

  #pragma unroll
  for (int dt = 0; dt < 8; ++dt) {
    v8f acc = {};
    const int dcol = dt * 16 + lmod;
    #pragma unroll
    for (int kk = 0; kk < 32; ++kk) {
      const int krow = kk * 4 + lhalf * 2;
      v2f bfrag = (v2f){Wl[krow * D_MP + dcol], Wl[(krow + 1) * D_MP + dcol]};
      acc = wmma_f32_4(afrag[kk], bfrag, acc);
    }
    const float bias = bl[dcol];
    #pragma unroll
    for (int r = 0; r < 8; ++r) {
      const int row = m0 + r + 8 * lhalf;   // C/D: VGPR r -> M=r (+8 hi half)
      float x = acc[r] + bias;
      float s = x / (1.0f + expf(-x));      // swish
      __bf16 hi, lo;
      bf16_split(s, hi, lo);
      tf_hiT[(size_t)dcol * N_CARDS + row] = hi;
      tf_loT[(size_t)dcol * N_CARDS + row] = lo;
    }
  }
}

// ---------------------------------------------------------------------------
// Kernel 2: path_embeddings = metapath^T @ transformed  [N_PATHS, D_MP]
// fp32 via two-term bf16 split: ah*bh + ah*bl + al*bh, fp32 WMMA accumulate.
// grid = N_PATHS/128 blocks of 256 threads; K = N_CARDS in chunks of 32.
// ---------------------------------------------------------------------------
#define KC     32
#define ROW_BF 40   // padded LDS row: 32 bf16 + 8 pad = 80 B (16B aligned,
                    // 20-dword bank stride -> conflict-free row readers)

__global__ __launch_bounds__(256) void mpe_path_gemm(
    const float* __restrict__ MP, const __bf16* __restrict__ BhT,
    const __bf16* __restrict__ BlT, float* __restrict__ PE) {
  __shared__ __bf16 Ah[128 * ROW_BF];  // metapath tile, [p][k], hi
  __shared__ __bf16 Al[128 * ROW_BF];  // metapath tile, [p][k], lo
  __shared__ __bf16 Bh[128 * ROW_BF];  // transformed tile, [d][k], hi
  __shared__ __bf16 Bl[128 * ROW_BF];  // transformed tile, [d][k], lo

  const int tid   = threadIdx.x;
  const int wave  = tid >> 5;
  const int lane  = tid & 31;
  const int lhalf = lane >> 4;
  const int lmod  = lane & 15;
  const int P0 = blockIdx.x * 128;

  v8f acc[8];
  #pragma unroll
  for (int dt = 0; dt < 8; ++dt) acc[dt] = (v8f){};

  const float4* mp4 = (const float4*)MP;

  for (int k0 = 0; k0 < N_CARDS; k0 += KC) {
    // ---- stage B (pure copy, pre-split & pre-transposed by kernel 1) ----
    // 1024 x 16B chunks: [hi|lo] x 128 d-rows x 4 quarters
    #pragma unroll
    for (int j = 0; j < 4; ++j) {
      const int idx = tid + j * 256;        // 0..1023
      const int arr = idx >> 9;             // 0 = hi, 1 = lo
      const int rem = idx & 511;
      const int d   = rem >> 2;             // 0..127
      const int q   = rem & 3;              // 16B quarter of the 64B row chunk
      const __bf16* src = (arr ? BlT : BhT) + (size_t)d * N_CARDS + k0 + q * 8;
      __bf16* dst = (arr ? Bl : Bh) + d * ROW_BF + q * 8;
      *(v8bf*)dst = *(const v8bf*)src;
    }
    // ---- stage A (convert fp32 -> bf16 hi/lo, transpose [k][p] -> [p][k]) --
    #pragma unroll
    for (int j = 0; j < 4; ++j) {
      const int idx = tid + j * 256;        // 0..1023 float4s
      const int r  = idx >> 5;              // k row 0..31
      const int c4 = idx & 31;              // float4 column
      const float4 f = mp4[(size_t)(k0 + r) * (N_PATHS / 4) + (P0 >> 2) + c4];
      const int p = c4 * 4;
      __bf16 hi, lo;
      bf16_split(f.x, hi, lo); Ah[(p + 0) * ROW_BF + r] = hi; Al[(p + 0) * ROW_BF + r] = lo;
      bf16_split(f.y, hi, lo); Ah[(p + 1) * ROW_BF + r] = hi; Al[(p + 1) * ROW_BF + r] = lo;
      bf16_split(f.z, hi, lo); Ah[(p + 2) * ROW_BF + r] = hi; Al[(p + 2) * ROW_BF + r] = lo;
      bf16_split(f.w, hi, lo); Ah[(p + 3) * ROW_BF + r] = hi; Al[(p + 3) * ROW_BF + r] = lo;
    }
    __syncthreads();

    // Prefetch next metapath chunk while computing this one.
    if (k0 + KC < N_CARDS && tid < 128) {
      const int pr = tid >> 2;
      const int pc = (tid & 3) * 32;
      __builtin_prefetch(&MP[(size_t)(k0 + KC + pr) * N_PATHS + P0 + pc], 0, 0);
    }

    // ---- compute: 16-bit A 16x32 layout: lane holds M=l&15,
    //      K runs [8h,8h+8) in VGPR0-3 and [16+8h,16+8h+8) in VGPR4-7 ----
    const __bf16* arow_h = Ah + (wave * 16 + lmod) * ROW_BF + 8 * lhalf;
    const __bf16* arow_l = Al + (wave * 16 + lmod) * ROW_BF + 8 * lhalf;
    const v16bf ah = cat8(*(const v8bf*)arow_h, *(const v8bf*)(arow_h + 16));
    const v16bf al = cat8(*(const v8bf*)arow_l, *(const v8bf*)(arow_l + 16));

    #pragma unroll
    for (int dt = 0; dt < 8; ++dt) {
      // B 32x16 layout: lane holds N=l&15, K = [8h,8h+8) then [16+8h,16+8h+8).
      const __bf16* brow_h = Bh + (dt * 16 + lmod) * ROW_BF + 8 * lhalf;
      const __bf16* brow_l = Bl + (dt * 16 + lmod) * ROW_BF + 8 * lhalf;
      const v16bf bh = cat8(*(const v8bf*)brow_h, *(const v8bf*)(brow_h + 16));
      const v16bf bl = cat8(*(const v8bf*)brow_l, *(const v8bf*)(brow_l + 16));
      acc[dt] = wmma_bf16_32(ah, bh, acc[dt]);
      acc[dt] = wmma_bf16_32(ah, bl, acc[dt]);
      acc[dt] = wmma_bf16_32(al, bh, acc[dt]);
    }
    __syncthreads();
  }

  #pragma unroll
  for (int dt = 0; dt < 8; ++dt) {
    #pragma unroll
    for (int r = 0; r < 8; ++r) {
      const int p = P0 + wave * 16 + r + 8 * lhalf;
      PE[(size_t)p * D_MP + dt * 16 + lmod] = acc[dt][r];
    }
  }
}

// ---------------------------------------------------------------------------
// Kernel 3: zero the output accumulator
// ---------------------------------------------------------------------------
__global__ void mpe_zero_out(float4* __restrict__ out, int n4) {
  const int i = blockIdx.x * blockDim.x + threadIdx.x;
  if (i < n4) out[i] = (float4){0.f, 0.f, 0.f, 0.f};
}

// ---------------------------------------------------------------------------
// Kernel 4: sparse pool scatter. One wave per 32 sorted-nnz chunk; lanes cover
// D_MP as float4; register-accumulate same-row runs, flush with f32 atomics.
// ---------------------------------------------------------------------------
__global__ __launch_bounds__(256) void mpe_pool_scatter(
    const float* __restrict__ PE, const int* __restrict__ prow,
    const int* __restrict__ pcol, const float* __restrict__ pval,
    float* __restrict__ out) {
  const int gw   = (blockIdx.x * 256 + threadIdx.x) >> 5;
  const int lane = threadIdx.x & 31;
  const int start = gw * 32;
  if (start >= NNZ_TOT) return;
  int end = start + 32;
  if (end > NNZ_TOT) end = NNZ_TOT;

  const float4* pe4 = (const float4*)PE;
  float4 acc = {0.f, 0.f, 0.f, 0.f};
  int cur = -1;

  for (int i = start; i < end; ++i) {
    const int r = prow[i];
    const int c = pcol[i];
    const float v = pval[i];
    if (r != cur) {
      if (cur >= 0) {
        float* dst = out + (size_t)cur * D_MP + lane * 4;
        atomicAdd(dst + 0, acc.x);
        atomicAdd(dst + 1, acc.y);
        atomicAdd(dst + 2, acc.z);
        atomicAdd(dst + 3, acc.w);
      }
      acc = (float4){0.f, 0.f, 0.f, 0.f};
      cur = r;
    }
    const float4 g = pe4[(size_t)c * (D_MP / 4) + lane];
    acc.x += g.x * v;
    acc.y += g.y * v;
    acc.z += g.z * v;
    acc.w += g.w * v;
  }
  if (cur >= 0) {
    float* dst = out + (size_t)cur * D_MP + lane * 4;
    atomicAdd(dst + 0, acc.x);
    atomicAdd(dst + 1, acc.y);
    atomicAdd(dst + 2, acc.z);
    atomicAdd(dst + 3, acc.w);
  }
}

// ---------------------------------------------------------------------------
extern "C" void kernel_launch(void* const* d_in, const int* in_sizes, int n_in,
                              void* d_out, int out_size, void* d_ws, size_t ws_size,
                              hipStream_t stream) {
  const float* E    = (const float*)d_in[0];  // card_embeddings [16384,128]
  const float* MP   = (const float*)d_in[1];  // metapath        [16384,16384]
  const float* W    = (const float*)d_in[2];  // [128,128]
  const float* b    = (const float*)d_in[3];  // [128]
  const int*   prow = (const int*)d_in[4];    // [NNZ]
  const int*   pcol = (const int*)d_in[5];    // [NNZ]
  const float* pval = (const float*)d_in[6];  // [NNZ]
  float* out = (float*)d_out;                 // [4096,128]

  // Workspace: tf_hiT (4 MB) | tf_loT (4 MB) | path_embeddings (8 MB)
  __bf16* tf_hiT = (__bf16*)d_ws;
  __bf16* tf_loT = tf_hiT + (size_t)D_MP * N_CARDS;
  float*  pe     = (float*)(tf_loT + (size_t)D_MP * N_CARDS);

  mpe_transform_swish<<<N_CARDS / 128, 256, 0, stream>>>(E, W, b, tf_hiT, tf_loT);
  mpe_path_gemm<<<N_PATHS / 128, 256, 0, stream>>>(MP, tf_hiT, tf_loT, pe);

  const int n4 = BATCHSZ * D_MP / 4;
  mpe_zero_out<<<(n4 + 255) / 256, 256, 0, stream>>>((float4*)out, n4);

  const int nwaves = NNZ_TOT / 32;
  mpe_pool_scatter<<<(nwaves * 32) / 256, 256, 0, stream>>>(pe, prow, pcol, pval, out);
}
